// BrainTumorGCNN_27290222198847
// MI455X (gfx1250) — matrix operand order repaired and checked by reading, not compiled, not run on hardware
//
#include <hip/hip_runtime.h>
#include <hip/hip_bf16.h>
#include <math.h>

// ---------------------------------------------------------------------------
// CDNA5 (gfx1250) WMMA / vector types
// ---------------------------------------------------------------------------
typedef __attribute__((ext_vector_type(16))) __bf16 v16bf;
typedef __attribute__((ext_vector_type(8)))  __bf16 v8bf;
typedef __attribute__((ext_vector_type(4)))  __bf16 v4bf;
typedef __attribute__((ext_vector_type(8)))  float  v8f;
typedef __attribute__((ext_vector_type(4)))  float  v4f;
typedef __attribute__((ext_vector_type(4)))  unsigned int u32x4;
typedef __attribute__((ext_vector_type(8)))  int    i32x8;
typedef __attribute__((ext_vector_type(4)))  int    i32x4;

static __device__ __forceinline__ v8f wmma_bf16(v16bf a, v16bf b, v8f c) {
    return __builtin_amdgcn_wmma_f32_16x16x32_bf16(
        false, a, false, b, (short)0, c, false, false);
}

// ---------------------------------------------------------------------------
// Tensor Data Mover staging: 1-D copy (nbytes, multiple of 8, n64 < 65536)
// expressed as a 2-D u64 tensor per CDNA5 ISA.
// lds byte offset relative to workgroup LDS base (single static __shared__
// array -> offset 0). Fallback: plain vector copy.
// ---------------------------------------------------------------------------
static __device__ __forceinline__ bool tdm_available() {
#if defined(__has_builtin) && __has_builtin(__builtin_amdgcn_tensor_load_to_lds)
    return true;
#else
    return false;
#endif
}

static __device__ __forceinline__ void tdm_load_to_lds(const void* gsrc,
                                                       unsigned ldsByteOff,
                                                       unsigned nbytes) {
#if defined(__has_builtin) && __has_builtin(__builtin_amdgcn_tensor_load_to_lds)
    const unsigned n64 = nbytes >> 3;              // u64 elements
    const unsigned long long ga = (unsigned long long)gsrc;
    u32x4 g0;
    g0[0] = 1u;                                    // count=1, user descriptor
    g0[1] = ldsByteOff;                            // lds_addr (bytes)
    g0[2] = (unsigned)ga;                          // global_addr[31:0]
    g0[3] = (unsigned)(ga >> 32) | (2u << 30);     // global_addr[56:32] | type=2
    i32x8 g1;
    g1[0] = (int)(3u << 16);                       // wg_mask=0, data_size=3 (8B)
    g1[1] = (int)(n64 << 16);                      // tensor_dim0[15:0]
    g1[2] = (int)((n64 >> 16) | (1u << 16));       // tensor_dim0[31:16] | tensor_dim1=1
    g1[3] = (int)(n64 << 16);                      // tile_dim0 = n64
    g1[4] = 0;                                     // tile_dim1=0, tile_dim2=0
    g1[5] = (int)n64;                              // tensor_dim0_stride[31:0]
    g1[6] = 0;
    g1[7] = 0;
    i32x4 z4 = {0, 0, 0, 0};
#if __clang_major__ >= 23
    i32x8 z8 = {0, 0, 0, 0, 0, 0, 0, 0};
    __builtin_amdgcn_tensor_load_to_lds(g0, g1, z4, z4, z8, 0);
#else
    __builtin_amdgcn_tensor_load_to_lds(g0, g1, z4, z4, 0);
#endif
    __builtin_amdgcn_s_wait_tensorcnt(0);
#else
    (void)gsrc; (void)ldsByteOff; (void)nbytes;
#endif
}

// Cooperative fallback copy (global -> LDS), 16B per thread per iteration.
static __device__ __forceinline__ void sync_copy_to_lds(__bf16* lds,
                                                        const __bf16* gsrc,
                                                        int elems) {
    for (int i = threadIdx.x; i < (elems >> 3); i += (int)blockDim.x)
        *(v8bf*)(lds + i * 8) = *(const v8bf*)(gsrc + i * 8);
}

// ---------------------------------------------------------------------------
// Fragment loaders (CDNA5 ISA §7.12.2, wave32)
// A (16x32 bf16): lane L -> row M=L&15; contiguous run [kb,kb+16)
// with kb = 16*(L>=16). Fully vectorial (load -> convertvector
// -> shuffle) so the optimizer cannot narrow the b128 loads to scalars.
// ---------------------------------------------------------------------------
static __device__ __forceinline__ v16bf load_a_frag_f32(const float* src, long ld) {
    const int lane = threadIdx.x & 31;
    const int m    = lane & 15;
    const int kb   = (lane >> 4) << 4;
    const float* row = src + (long)m * ld + kb;
    v4f a0 = *(const v4f*)(row);
    v4f a1 = *(const v4f*)(row + 4);
    v4f a2 = *(const v4f*)(row + 8);
    v4f a3 = *(const v4f*)(row + 12);
    v4bf c0 = __builtin_convertvector(a0, v4bf);
    v4bf c1 = __builtin_convertvector(a1, v4bf);
    v4bf c2 = __builtin_convertvector(a2, v4bf);
    v4bf c3 = __builtin_convertvector(a3, v4bf);
    v8bf lo = __builtin_shufflevector(c0, c1, 0, 1, 2, 3, 4, 5, 6, 7);
    v8bf hi = __builtin_shufflevector(c2, c3, 0, 1, 2, 3, 4, 5, 6, 7);
    return __builtin_shufflevector(lo, hi, 0, 1, 2, 3, 4, 5, 6, 7,
                                   8, 9, 10, 11, 12, 13, 14, 15);
}

static __device__ __forceinline__ v16bf load_a_frag_bf16(const __bf16* src, long ld) {
    const int lane = threadIdx.x & 31;
    const int m    = lane & 15;
    const int kb   = (lane >> 4) << 4;
    const __bf16* row = src + (long)m * ld + kb;
    v8bf lo = *(const v8bf*)(row);
    v8bf hi = *(const v8bf*)(row + 8);
    return __builtin_shufflevector(lo, hi, 0, 1, 2, 3, 4, 5, 6, 7,
                                   8, 9, 10, 11, 12, 13, 14, 15);
}

// 8-row A matrix padded to 16 (rows 8..15 mirror 0..7, discarded at store)
static __device__ __forceinline__ v16bf load_a_frag_bf16_rows8(const __bf16* src, long ld) {
    const int lane = threadIdx.x & 31;
    const int m    = lane & 7;
    const int kb   = (lane >> 4) << 4;
    const __bf16* row = src + (long)m * ld + kb;
    v8bf lo = *(const v8bf*)(row);
    v8bf hi = *(const v8bf*)(row + 8);
    return __builtin_shufflevector(lo, hi, 0, 1, 2, 3, 4, 5, 6, 7,
                                   8, 9, 10, 11, 12, 13, 14, 15);
}

// B (32x16) from row-major KxN fp32 (weights): strided scalar loads.
static __device__ __forceinline__ v16bf load_b_frag_f32(const float* src, long ld) {
    const int lane = threadIdx.x & 31;
    const int n    = lane & 15;
    const int kb   = (lane >> 4) << 4;
    v16bf b;
#pragma unroll
    for (int j = 0; j < 16; ++j)
        b[j] = (__bf16)src[(long)(kb + j) * ld + n];
    return b;
}

// B fragment from swizzled layout: per 32x16 tile, lane L owns 16 contiguous
// bf16 (position p holds K = 16*(L>=16) + p, column N = L&15).
// Works for both global ws and LDS pointers.
static __device__ __forceinline__ v16bf load_b_frag_swz(const __bf16* tile) {
    const __bf16* p = tile + (threadIdx.x & 31) * 16;
    v8bf lo = *(const v8bf*)(p);
    v8bf hi = *(const v8bf*)(p + 8);
    return __builtin_shufflevector(lo, hi, 0, 1, 2, 3, 4, 5, 6, 7,
                                   8, 9, 10, 11, 12, 13, 14, 15);
}

// Store a 16x16 C/D tile into the swizzled B layout of a 32x16 K-tile.
// upper = 1 if this tile is rows 16..31 of the K-tile. One b128 store/lane.
static __device__ __forceinline__ void store_tile_bfrag(__bf16* tile, int upper, v8f c) {
    const int lane = threadIdx.x & 31;
    const int n    = lane & 15;
    const int mb   = (lane >> 4) << 3;
    v8bf v = __builtin_convertvector(c, v8bf);
    *(v8bf*)(tile + (n + 16 * upper) * 16 + mb) = v;
}

// Row-major C/D store with fused bias+relu (for A-side consumers).
static __device__ __forceinline__ void store_tile_relu_bias_bf16(
        __bf16* dst, long ld, v8f c, const float* bias, int n0) {
    const int lane = threadIdx.x & 31;
    const int n    = lane & 15;
    const int mb   = (lane >> 4) << 3;
    const float bn = bias[n0 + n];
#pragma unroll
    for (int j = 0; j < 8; ++j) {
        float v = c[j] + bn;
        dst[(long)(mb + j) * ld + n] = (__bf16)(v > 0.f ? v : 0.f);
    }
}

// ---------------------------------------------------------------------------
// Problem constants
// ---------------------------------------------------------------------------
#define BB 8
#define NN 2048
#define FF 128
#define H1C 32
#define H2C 64
#define D1C 128
#define ROWS (BB * NN)        // 16384
#define KT_PER_B (NN / 32)    // 64 K-tiles per batch panel

// ---------------------------------------------------------------------------
// K1: XW1 = x @ W1 -> swizzled B panel for agg1
// ---------------------------------------------------------------------------
__global__ void __launch_bounds__(256) k_xw1(const float* __restrict__ x,
                                             const float* __restrict__ W1,
                                             __bf16* __restrict__ xw1swz) {
    const int wave = blockIdx.x * 8 + (threadIdx.x >> 5);
    const long m0  = (long)wave * 16;
    const float* xr = x + m0 * FF;
    v8f acc0 = {}, acc1 = {};
#pragma unroll
    for (int kk = 0; kk < FF; kk += 32) {
        v16bf af  = load_a_frag_f32(xr + kk, FF);
        v16bf bf0 = load_b_frag_f32(W1 + (long)kk * H1C, H1C);
        v16bf bf1 = load_b_frag_f32(W1 + (long)kk * H1C + 16, H1C);
        acc0 = wmma_bf16(af, bf0, acc0);
        acc1 = wmma_bf16(af, bf1, acc1);
    }
    const int upper = (int)((m0 >> 4) & 1);
    __bf16* base = xw1swz + ((m0 >> 5) * 2) * 512;
    store_tile_bfrag(base,       upper, acc0);
    store_tile_bfrag(base + 512, upper, acc1);
}

// ---------------------------------------------------------------------------
// K2: h1 = relu(A @ XW1 + b1). B panel (128 KB swizzled) staged in LDS via
// TDM; A streamed fp32 with vectorized loads + prefetch.
// ---------------------------------------------------------------------------
__global__ void __launch_bounds__(256) k_agg1(const float* __restrict__ A,
                                              const __bf16* __restrict__ xw1swz,
                                              const float* __restrict__ bias1,
                                              __bf16* __restrict__ h1) {
    __shared__ __bf16 sB[KT_PER_B * 2 * 512];          // 128 KB
    const int wave = blockIdx.x * 8 + (threadIdx.x >> 5);
    const int b    = wave >> 7;
    const int rt   = wave & 127;
    const int lane = threadIdx.x & 31;

    const __bf16* gB = xw1swz + (long)b * KT_PER_B * 2 * 512;
    if (tdm_available()) {
        if ((threadIdx.x >> 5) == 0)
            tdm_load_to_lds(gB, 0, KT_PER_B * 2 * 512 * 2);
    } else {
        sync_copy_to_lds(sB, gB, KT_PER_B * 2 * 512);
    }
    __syncthreads();

    const float* Ab = A + (long)b * NN * NN + (long)rt * 16 * NN;
    v8f acc0 = {}, acc1 = {};
    for (int kk = 0; kk < NN; kk += 32) {
        if (kk + 256 < NN)
            __builtin_prefetch(Ab + (long)(lane & 15) * NN + kk + 256, 0, 1);
        v16bf af = load_a_frag_f32(Ab + kk, NN);
        const __bf16* tile = sB + ((kk >> 5) * 2) * 512;
        acc0 = wmma_bf16(af, load_b_frag_swz(tile), acc0);
        acc1 = wmma_bf16(af, load_b_frag_swz(tile + 512), acc1);
    }
    __bf16* out = h1 + (long)b * NN * H1C + (long)rt * 16 * H1C;
    store_tile_relu_bias_bf16(out,      H1C, acc0, bias1, 0);
    store_tile_relu_bias_bf16(out + 16, H1C, acc1, bias1, 16);
}

// ---------------------------------------------------------------------------
// K3: HW2 = h1 @ W2 -> swizzled B panel for agg2 (single K=32 step)
// ---------------------------------------------------------------------------
__global__ void __launch_bounds__(256) k_hw2(const __bf16* __restrict__ h1,
                                             const float* __restrict__ W2,
                                             __bf16* __restrict__ hw2swz) {
    const int wave = blockIdx.x * 8 + (threadIdx.x >> 5);
    const long m0  = (long)wave * 16;
    const int upper = (int)((m0 >> 4) & 1);
    v16bf af = load_a_frag_bf16(h1 + m0 * H1C, H1C);
    __bf16* base = hw2swz + ((m0 >> 5) * 4) * 512;
#pragma unroll
    for (int t = 0; t < 4; ++t) {
        v8f acc = {};
        acc = wmma_bf16(af, load_b_frag_f32(W2 + t * 16, H2C), acc);
        store_tile_bfrag(base + t * 512, upper, acc);
    }
}

// ---------------------------------------------------------------------------
// K4: h2 = relu(A @ HW2 + b2). 256 KB B panel staged in two 128 KB phases.
// ---------------------------------------------------------------------------
__global__ void __launch_bounds__(256) k_agg2(const float* __restrict__ A,
                                              const __bf16* __restrict__ hw2swz,
                                              const float* __restrict__ bias2,
                                              __bf16* __restrict__ h2) {
    __shared__ __bf16 sB[32 * 4 * 512];                // 128 KB (half panel)
    const int wave = blockIdx.x * 8 + (threadIdx.x >> 5);
    const int b    = wave >> 7;
    const int rt   = wave & 127;
    const int lane = threadIdx.x & 31;

    const float* Ab = A + (long)b * NN * NN + (long)rt * 16 * NN;
    v8f acc[4] = {{}, {}, {}, {}};

    for (int ph = 0; ph < 2; ++ph) {
        if (ph) __syncthreads();                       // drain readers
        const __bf16* gB = hw2swz + ((long)b * KT_PER_B + ph * 32) * 4 * 512;
        if (tdm_available()) {
            if ((threadIdx.x >> 5) == 0)
                tdm_load_to_lds(gB, 0, 32 * 4 * 512 * 2);
        } else {
            sync_copy_to_lds(sB, gB, 32 * 4 * 512);
        }
        __syncthreads();

        const int kend = ph * 1024 + 1024;
        for (int kk = ph * 1024; kk < kend; kk += 32) {
            if (kk + 256 < NN)
                __builtin_prefetch(Ab + (long)(lane & 15) * NN + kk + 256, 0, 1);
            v16bf af = load_a_frag_f32(Ab + kk, NN);
            const __bf16* tile = sB + (((kk >> 5) - ph * 32) * 4) * 512;
#pragma unroll
            for (int t = 0; t < 4; ++t)
                acc[t] = wmma_bf16(af, load_b_frag_swz(tile + t * 512), acc[t]);
        }
    }
    __bf16* out = h2 + (long)b * NN * H2C + (long)rt * 16 * H2C;
#pragma unroll
    for (int t = 0; t < 4; ++t)
        store_tile_relu_bias_bf16(out + t * 16, H2C, acc[t], bias2, t * 16);
}

// ---------------------------------------------------------------------------
// K5: dense partials [8 x 131072] @ [131072 x 128]; K split 64 ways; f32
// atomics combine partial tiles.
// ---------------------------------------------------------------------------
#define KTOT (NN * H2C)       // 131072
#define KCHUNK 2048

__global__ void __launch_bounds__(256) k_dense(const __bf16* __restrict__ h2flat,
                                               const float* __restrict__ Wd,
                                               float* __restrict__ acc) {
    const int wave   = blockIdx.x * 8 + (threadIdx.x >> 5);
    const int ntile  = wave & 7;
    const int kchunk = wave >> 3;
    const long k0    = (long)kchunk * KCHUNK;
    const int  n0    = ntile * 16;
    const int  lane  = threadIdx.x & 31;
    v8f c = {};
    for (int kk = 0; kk < KCHUNK; kk += 32) {
        if (kk + 256 < KCHUNK)   // prefetch the Wd stream 8 K-tiles ahead
            __builtin_prefetch(Wd + (k0 + kk + 256 + lane) * D1C + n0, 0, 1);
        v16bf af = load_a_frag_bf16_rows8(h2flat + k0 + kk, (long)KTOT);
        v16bf bf = load_b_frag_f32(Wd + (k0 + kk) * D1C + n0, D1C);
        c = wmma_bf16(af, bf, c);
    }
    const int n  = lane & 15;
    const int mb = (lane >> 4) << 3;
#pragma unroll
    for (int j = 0; j < 8; ++j) {
        const int m = mb + j;
        if (m < BB) atomicAdd(&acc[m * D1C + n0 + n], c[j]);
    }
}

// ---------------------------------------------------------------------------
// K6: out[b] = sigmoid(relu(acc[b,:] + bd) . Wo + bo)
// ---------------------------------------------------------------------------
__global__ void k_final(const float* __restrict__ acc,
                        const float* __restrict__ bd,
                        const float* __restrict__ Wo,
                        const float* __restrict__ bo,
                        float* __restrict__ out) {
    __shared__ float red[D1C];
    const int b = blockIdx.x;
    const int t = threadIdx.x;
    float h = acc[b * D1C + t] + bd[t];
    h = h > 0.f ? h : 0.f;
    red[t] = h * Wo[t];
    __syncthreads();
#pragma unroll
    for (int s = D1C / 2; s > 0; s >>= 1) {
        if (t < s) red[t] += red[t + s];
        __syncthreads();
    }
    if (t == 0) out[b] = 1.f / (1.f + expf(-(red[0] + bo[0])));
}

// ---------------------------------------------------------------------------
// Launch
// ---------------------------------------------------------------------------
extern "C" void kernel_launch(void* const* d_in, const int* in_sizes, int n_in,
                              void* d_out, int out_size, void* d_ws, size_t ws_size,
                              hipStream_t stream) {
    (void)in_sizes; (void)n_in; (void)out_size; (void)ws_size;
    const float* x   = (const float*)d_in[0];
    const float* a   = (const float*)d_in[1];
    const float* W1  = (const float*)d_in[2];
    const float* b1  = (const float*)d_in[3];
    const float* W2  = (const float*)d_in[4];
    const float* b2  = (const float*)d_in[5];
    const float* Wd  = (const float*)d_in[6];
    const float* bd  = (const float*)d_in[7];
    const float* Wo  = (const float*)d_in[8];
    const float* bo  = (const float*)d_in[9];
    float* out = (float*)d_out;

    char* ws = (char*)d_ws;
    __bf16* xw1swz = (__bf16*)(ws);                 // 1 MB swizzled
    __bf16* h1     = (__bf16*)(ws + (1u << 20));    // 1 MB row-major
    __bf16* hw2swz = (__bf16*)(ws + (2u << 20));    // 2 MB swizzled
    __bf16* h2     = (__bf16*)(ws + (4u << 20));    // 2 MB row-major
    float*  acc    = (float*)(ws + (6u << 20));     // 4 KB

    hipMemsetAsync(acc, 0, BB * D1C * sizeof(float), stream);

    k_xw1  <<<ROWS / 16 / 8, 256, 0, stream>>>(x, W1, xw1swz);
    k_agg1 <<<ROWS / 16 / 8, 256, 0, stream>>>(a, xw1swz, b1, h1);
    k_hw2  <<<ROWS / 16 / 8, 256, 0, stream>>>(h1, W2, hw2swz);
    k_agg2 <<<ROWS / 16 / 8, 256, 0, stream>>>(a, hw2swz, b2, h2);
    k_dense<<<64,            256, 0, stream>>>(h2, Wd, acc);
    k_final<<<BB,            D1C, 0, stream>>>(acc, bd, Wo, bo, out);
}
